// KANBlock_4801773437391
// MI455X (gfx1250) — matrix operand order repaired and checked by reading, not compiled, not run on hardware
//
#include <hip/hip_runtime.h>
#include <hip/hip_bf16.h>

// ---------------- problem constants ----------------
#define H_    1024
#define NH_   16
#define HD_   64
#define S_    1024
#define B_    2
#define T_    (B_*S_)      // 2048 tokens
#define DFF_  1024
#define E_    4
#define NB_   8            // G + K_ORD = 5 + 3
#define KF_   (H_*(NB_+1)) // 9216 combined feature dim (base + 8 splines)

typedef __attribute__((ext_vector_type(16))) __bf16 v16bf;
typedef __attribute__((ext_vector_type(8)))  float  v8f;

// ---------------- helpers ----------------
static __device__ __forceinline__ unsigned short f2bf(float f){
  union { float f; unsigned u; } v; v.f = f;
  unsigned u = v.u;
  u += 0x7FFFu + ((u >> 16) & 1u);   // round-to-nearest-even
  return (unsigned short)(u >> 16);
}
static __device__ __forceinline__ __bf16 us2bf(unsigned short s){
  union { unsigned short u; __bf16 b; } v; v.u = s; return v.b;
}
static __device__ __forceinline__ float silu_f(float x){ return x / (1.0f + __expf(-x)); }

// Cox-de Boor, G=5, K=3 -> 8 basis functions, grid [-2.2 .. 2.2] step 0.4
static __device__ __forceinline__ void bspl8(float x, float* o){
  const float h = 2.0f / 5.0f;
  float g[12];
  #pragma unroll
  for (int j = 0; j < 12; ++j) g[j] = (float)(j - 3) * h - 1.0f;
  float b[11];
  #pragma unroll
  for (int j = 0; j < 11; ++j) b[j] = (x >= g[j] && x < g[j+1]) ? 1.0f : 0.0f;
  #pragma unroll
  for (int k = 1; k <= 3; ++k){
    #pragma unroll
    for (int j = 0; j < 11 - k; ++j){
      float l = (x - g[j])     / (g[j+k]   - g[j])   * b[j];
      float r = (g[j+k+1] - x) / (g[j+k+1] - g[j+1]) * b[j+1];
      b[j] = l + r;
    }
  }
  #pragma unroll
  for (int j = 0; j < 8; ++j) o[j] = b[j];
}

// ---------------- rmsnorm + KAN feature build ----------------
__global__ __launch_bounds__(256) void k_rmsnorm_feat(
    const float* __restrict__ x, const float* __restrict__ w,
    unsigned short* __restrict__ F, float* __restrict__ hn)
{
  __shared__ float red[256];
  const int t = blockIdx.x, tid = threadIdx.x;
  const float* xr = x + (size_t)t * H_;
  float s = 0.f;
  for (int i = tid; i < H_; i += 256){ float v = xr[i]; s += v * v; }
  red[tid] = s; __syncthreads();
  for (int off = 128; off > 0; off >>= 1){
    if (tid < off) red[tid] += red[tid + off];
    __syncthreads();
  }
  const float inv = rsqrtf(red[0] / (float)H_ + 1e-6f);
  unsigned short* Fr = F + (size_t)t * KF_;
  for (int i = tid; i < H_; i += 256){
    float v = xr[i] * inv * w[i];
    if (hn) hn[(size_t)t * H_ + i] = v;
    float bs[8]; bspl8(v, bs);
    unsigned short* p = Fr + (size_t)i * 9;
    p[0] = f2bf(silu_f(v));
    #pragma unroll
    for (int c = 0; c < 8; ++c) p[1 + c] = f2bf(bs[c]);
  }
}

// ---------------- KAN weight fuse+convert: W[(i*9+c)][o] (bf16, K-major) ----------------
__global__ __launch_bounds__(256) void k_conv_kan_w(
    const float* __restrict__ bw, const float* __restrict__ sw,
    const float* __restrict__ sc, unsigned short* __restrict__ W, int O, int I)
{
  int idx = blockIdx.x * 256 + threadIdx.x;
  if (idx >= O * I) return;
  const int o = idx / I, i = idx % I;
  const size_t wbase = ((size_t)i * 9) * O + o;
  W[wbase] = f2bf(bw[(size_t)o * I + i]);
  const float scal = sc[(size_t)o * I + i];
  const float* sp = sw + ((size_t)o * I + i) * NB_;
  #pragma unroll
  for (int c = 0; c < 8; ++c) W[wbase + (size_t)(1 + c) * O] = f2bf(sp[c] * scal);
}

// dense transpose+convert (for out_w):  W[i][o] = w[o][i]
__global__ __launch_bounds__(256) void k_conv_dense_T(
    const float* __restrict__ w, unsigned short* __restrict__ W, int O, int I)
{
  int idx = blockIdx.x * 256 + threadIdx.x;
  if (idx >= O * I) return;
  const int o = idx / I, i = idx % I;
  W[(size_t)i * O + o] = f2bf(w[(size_t)o * I + i]);
}

__global__ void k_f32_to_bf16(const float* __restrict__ a, unsigned short* __restrict__ b, int n){
  int i = blockIdx.x * blockDim.x + threadIdx.x;
  if (i < n) b[i] = f2bf(a[i]);
}

// ---------------- bf16 WMMA GEMM: C[MxN] f32 = A[MxK] @ B[KxN], row-major ----------------
// 256 threads = 8 waves; block tile 128x128, K-step 32; wave tile 32x64 = 2x4 WMMA tiles.
// Double-buffered LDS, filled with async global->LDS (ASYNCcnt) while WMMAs run.
template<int N, int K>
__global__ __launch_bounds__(256) void k_gemm_bf16(
    const unsigned short* __restrict__ A, const unsigned short* __restrict__ Bm,
    float* __restrict__ C)
{
  __shared__ unsigned short lA[2][128][40];   // 128x32 + pad, x2 buffers
  __shared__ unsigned short lB[2][32][144];   // 32x128 + pad, x2 buffers
  const int tid  = threadIdx.x;
  const int lane = tid & 31;
  const int wv   = tid >> 5;
  const int wm   = wv & 3;                 // 4 wave-rows
  const int wn   = wv >> 2;                // 2 wave-cols
  const int hlf  = lane >> 4;
  const int l16  = lane & 15;
  const int mBase = wm * 32;
  const int nBase = wn * 64;
  const int gM = blockIdx.y * 128;
  const int gN = blockIdx.x * 128;

  // per-thread staging coordinates: 2x 16B chunks for A tile, 2x for B tile
  int aRow[2], aCol[2], bRow[2], bCol[2];
  #pragma unroll
  for (int p = 0; p < 2; ++p){
    const int vec = tid + p * 256;          // 0..511
    aRow[p] = vec >> 2;  aCol[p] = (vec & 3) << 3;
    bRow[p] = vec >> 4;  bCol[p] = (vec & 15) << 3;
  }

  // async global->LDS stage of one 128x32 A tile + 32x128 B tile
  auto stage = [&](int buf, int k0){
    #pragma unroll
    for (int p = 0; p < 2; ++p){
      const unsigned la = (unsigned)(size_t)&lA[buf][aRow[p]][aCol[p]];
      const unsigned long long ga =
          (unsigned long long)(A + (size_t)(gM + aRow[p]) * K + k0 + aCol[p]);
      asm volatile("global_load_async_to_lds_b128 %0, %1, off"
                   :: "v"(la), "v"(ga) : "memory");
      const unsigned lb = (unsigned)(size_t)&lB[buf][bRow[p]][bCol[p]];
      const unsigned long long gb =
          (unsigned long long)(Bm + (size_t)(k0 + bRow[p]) * N + gN + bCol[p]);
      asm volatile("global_load_async_to_lds_b128 %0, %1, off"
                   :: "v"(lb), "v"(gb) : "memory");
    }
  };

  v8f acc[2][4];
  #pragma unroll
  for (int i = 0; i < 2; ++i)
    #pragma unroll
    for (int j = 0; j < 4; ++j)
      #pragma unroll
      for (int r = 0; r < 8; ++r) acc[i][j][r] = 0.0f;

  constexpr int kTiles = K >> 5;
  stage(0, 0);
  asm volatile("s_wait_asynccnt 0x0" ::: "memory");
  __syncthreads();

  int buf = 0;
  for (int kt = 0; kt < kTiles; ++kt){
    if (kt + 1 < kTiles) stage(buf ^ 1, (kt + 1) << 5);  // overlap with compute

    // A fragments (ISA 7.12.2: lanes hold M, K packed along VGPRs, halves split K)
    v16bf af[2];
    #pragma unroll
    for (int mi = 0; mi < 2; ++mi){
      const int row = mBase + mi * 16 + l16;
      #pragma unroll
      for (int v = 0; v < 8; ++v){
        const int kk = ((v >> 2) << 4) + (hlf << 3) + ((v & 3) << 1);
        af[mi][2 * v]     = us2bf(lA[buf][row][kk]);
        af[mi][2 * v + 1] = us2bf(lA[buf][row][kk + 1]);
      }
    }
    // B fragments (lanes hold N columns, K across vector elements, halves split K)
    v16bf bfr[4];
    #pragma unroll
    for (int ni = 0; ni < 4; ++ni){
      const int nb = nBase + ni * 16;
      #pragma unroll
      for (int e = 0; e < 16; ++e) bfr[ni][e] = us2bf(lB[buf][((e >> 3) << 4) + (hlf << 3) + (e & 7)][nb + l16]);
    }
    #pragma unroll
    for (int mi = 0; mi < 2; ++mi)
      #pragma unroll
      for (int ni = 0; ni < 4; ++ni)
        acc[mi][ni] = __builtin_amdgcn_wmma_f32_16x16x32_bf16(
            false, af[mi], false, bfr[ni], (short)0, acc[mi][ni], false, false);

    asm volatile("s_wait_asynccnt 0x0" ::: "memory");   // next tile landed in LDS
    __syncthreads();
    buf ^= 1;
  }

  // C/D layout: VGPR r -> row r + 8*half, col = lane&15
  #pragma unroll
  for (int mi = 0; mi < 2; ++mi)
    #pragma unroll
    for (int ni = 0; ni < 4; ++ni){
      const int col = gN + nBase + ni * 16 + l16;
      #pragma unroll
      for (int r = 0; r < 8; ++r){
        const int row = gM + mBase + mi * 16 + r + hlf * 8;
        C[(size_t)row * N + col] = acc[mi][ni][r];
      }
    }
}

// ---------------- RoPE + head split: qkv[T][3072] -> Q/K/V [B][NH][S][HD] ----------------
__global__ void k_rope_split(const float* __restrict__ qkv,
                             const float* __restrict__ rc, const float* __restrict__ rs,
                             float* __restrict__ Q, float* __restrict__ K, float* __restrict__ V)
{
  int idx = blockIdx.x * blockDim.x + threadIdx.x;   // B*NH*S*(HD/2)
  if (idx >= B_ * NH_ * S_ * (HD_ / 2)) return;
  const int d  = idx & 31;
  const int s  = (idx >> 5) & (S_ - 1);
  const int nh = (idx >> 15) & (NH_ - 1);
  const int b  = idx >> 19;
  const size_t rowq = ((size_t)(b * S_ + s)) * (3 * H_) + (size_t)nh * (3 * HD_);
  const float c  = rc[s * (HD_ / 2) + d];
  const float sn = rs[s * (HD_ / 2) + d];
  const size_t o = (((size_t)(b * NH_ + nh)) * S_ + s) * HD_ + 2 * d;
  const float qre = qkv[rowq + 2 * d],       qim = qkv[rowq + 2 * d + 1];
  const float kre = qkv[rowq + HD_ + 2 * d], kim = qkv[rowq + HD_ + 2 * d + 1];
  Q[o]     = qre * c - qim * sn;  Q[o + 1] = qre * sn + qim * c;
  K[o]     = kre * c - kim * sn;  K[o + 1] = kre * sn + kim * c;
  V[o]     = qkv[rowq + 2 * HD_ + 2 * d];
  V[o + 1] = qkv[rowq + 2 * HD_ + 2 * d + 1];
}

// ---------------- attention: one 64-thread block per (b,h,q) ----------------
__global__ __launch_bounds__(64) void k_attention(
    const float* __restrict__ Q, const float* __restrict__ K, const float* __restrict__ V,
    float* __restrict__ ctx)
{
  __shared__ float sc[S_];
  __shared__ float qv[HD_];
  __shared__ float red[64];
  const int tid = threadIdx.x;
  const int sq = blockIdx.x & (S_ - 1);
  const int h  = (blockIdx.x >> 10) & (NH_ - 1);
  const int b  = blockIdx.x >> 14;
  const size_t hb = ((size_t)(b * NH_ + h)) * S_ * HD_;
  qv[tid] = Q[hb + (size_t)sq * HD_ + tid];
  __syncthreads();
  for (int k = tid; k < S_; k += 64){
    const float* kr = K + hb + (size_t)k * HD_;
    float d = 0.f;
    #pragma unroll 8
    for (int j = 0; j < HD_; ++j) d += qv[j] * kr[j];
    sc[k] = d * 0.125f;                       // 1/sqrt(64)
  }
  __syncthreads();
  float m = -3.4e38f;
  for (int k = tid; k < S_; k += 64) m = fmaxf(m, sc[k]);
  red[tid] = m; __syncthreads();
  for (int off = 32; off > 0; off >>= 1){
    if (tid < off) red[tid] = fmaxf(red[tid], red[tid + off]);
    __syncthreads();
  }
  m = red[0]; __syncthreads();
  float s = 0.f;
  for (int k = tid; k < S_; k += 64){ float e = __expf(sc[k] - m); sc[k] = e; s += e; }
  red[tid] = s; __syncthreads();
  for (int off = 32; off > 0; off >>= 1){
    if (tid < off) red[tid] += red[tid + off];
    __syncthreads();
  }
  const float inv = 1.0f / red[0];
  float a = 0.f;
  for (int k = 0; k < S_; ++k) a += sc[k] * V[hb + (size_t)k * HD_ + tid];
  ctx[((size_t)(b * S_ + sq)) * H_ + h * HD_ + tid] = a * inv;
}

__global__ void k_add_bias_resid(const float* __restrict__ x, const float* __restrict__ proj,
                                 const float* __restrict__ bias, float* __restrict__ y, int n)
{
  int i = blockIdx.x * blockDim.x + threadIdx.x;
  if (i < n) y[i] = x[i] + proj[i] + bias[i % H_];
}

// ---------------- top-2 gating ----------------
__global__ __launch_bounds__(128) void k_gate(const float* __restrict__ h2,
                                              const float* __restrict__ gw,
                                              int* __restrict__ sel, float* __restrict__ sw)
{
  __shared__ float red[4][128];
  const int t = blockIdx.x, tid = threadIdx.x;
  const float* hr = h2 + (size_t)t * H_;
  float a0 = 0, a1 = 0, a2 = 0, a3 = 0;
  for (int i = tid; i < H_; i += 128){
    float v = hr[i];
    a0 += v * gw[0 * H_ + i]; a1 += v * gw[1 * H_ + i];
    a2 += v * gw[2 * H_ + i]; a3 += v * gw[3 * H_ + i];
  }
  red[0][tid] = a0; red[1][tid] = a1; red[2][tid] = a2; red[3][tid] = a3;
  __syncthreads();
  for (int off = 64; off > 0; off >>= 1){
    if (tid < off){
      red[0][tid] += red[0][tid + off]; red[1][tid] += red[1][tid + off];
      red[2][tid] += red[2][tid + off]; red[3][tid] += red[3][tid + off];
    }
    __syncthreads();
  }
  if (tid == 0){
    float l[4] = { red[0][0], red[1][0], red[2][0], red[3][0] };
    int i0 = 0;
    for (int e = 1; e < 4; ++e) if (l[e] > l[i0]) i0 = e;
    int i1 = -1;
    for (int e = 0; e < 4; ++e){ if (e == i0) continue; if (i1 < 0 || l[e] > l[i1]) i1 = e; }
    const float e1 = __expf(l[i1] - l[i0]);
    const float z = 1.0f + e1;
    sel[t * 2] = i0; sel[t * 2 + 1] = i1;
    sw[t * 2] = 1.0f / z; sw[t * 2 + 1] = e1 / z;
  }
}

// ---------------- z = t1*t2, then KAN features of z ----------------
__global__ __launch_bounds__(256) void k_mul_feat(const float* __restrict__ t1,
                                                  const float* __restrict__ t2,
                                                  unsigned short* __restrict__ F)
{
  const int t = blockIdx.x, tid = threadIdx.x;
  unsigned short* Fr = F + (size_t)t * KF_;
  for (int i = tid; i < DFF_; i += 256){
    float z = t1[(size_t)t * DFF_ + i] * t2[(size_t)t * DFF_ + i];
    float bs[8]; bspl8(z, bs);
    unsigned short* p = Fr + (size_t)i * 9;
    p[0] = f2bf(silu_f(z));
    #pragma unroll
    for (int c = 0; c < 8; ++c) p[1 + c] = f2bf(bs[c]);
  }
}

__global__ void k_copy(const float* __restrict__ a, float* __restrict__ b, int n){
  int i = blockIdx.x * blockDim.x + threadIdx.x;
  if (i < n) b[i] = a[i];
}

__global__ void k_moe_accum(const float* __restrict__ Ye, const int* __restrict__ sel,
                            const float* __restrict__ sw, int e, float* __restrict__ out, int n){
  int i = blockIdx.x * blockDim.x + threadIdx.x;
  if (i >= n) return;
  const int t = i / H_;
  const float w = (sel[t * 2] == e) ? sw[t * 2] : ((sel[t * 2 + 1] == e) ? sw[t * 2 + 1] : 0.0f);
  out[i] += w * Ye[i];
}

// ---------------- host orchestration ----------------
extern "C" void kernel_launch(void* const* d_in, const int* in_sizes, int n_in,
                              void* d_out, int out_size, void* d_ws, size_t ws_size,
                              hipStream_t stream)
{
  (void)in_sizes; (void)n_in; (void)out_size; (void)ws_size;
  const float* x        = (const float*)d_in[0];
  const float* rot_cos  = (const float*)d_in[1];
  const float* rot_sin  = (const float*)d_in[2];
  const float* norm1_w  = (const float*)d_in[3];
  const float* norm2_w  = (const float*)d_in[4];
  const float* qkv_base = (const float*)d_in[5];
  const float* qkv_spl  = (const float*)d_in[6];
  const float* qkv_scal = (const float*)d_in[7];
  const float* out_w    = (const float*)d_in[8];
  const float* out_b    = (const float*)d_in[9];
  const float* gate_w   = (const float*)d_in[10];
  const float* w1_base  = (const float*)d_in[11];
  const float* w1_spl   = (const float*)d_in[12];
  const float* w1_scal  = (const float*)d_in[13];
  const float* w2_base  = (const float*)d_in[14];
  const float* w2_spl   = (const float*)d_in[15];
  const float* w2_scal  = (const float*)d_in[16];
  const float* w3_base  = (const float*)d_in[17];
  const float* w3_spl   = (const float*)d_in[18];
  const float* w3_scal  = (const float*)d_in[19];
  float* out = (float*)d_out;

  char* ws = (char*)d_ws;
  size_t off = 0;
  auto alloc = [&](size_t bytes) -> char* {
    char* p = ws + off;
    off += (bytes + 255) & ~(size_t)255;
    return p;
  };
  unsigned short* F1   = (unsigned short*)alloc((size_t)T_ * KF_ * 2);        // reused as F3
  unsigned short* WB   = (unsigned short*)alloc((size_t)KF_ * (3 * H_) * 2);  // one weight buf
  float* qkvout        = (float*)alloc((size_t)T_ * (3 * H_) * 4);            // reused as T1,T2
  float* Qb            = (float*)alloc((size_t)T_ * H_ * 4);                  // reused as Ye
  float* Kb            = (float*)alloc((size_t)T_ * H_ * 4);
  float* Vb            = (float*)alloc((size_t)T_ * H_ * 4);
  float* ctx           = (float*)alloc((size_t)T_ * H_ * 4);
  unsigned short* ctxb = (unsigned short*)alloc((size_t)T_ * H_ * 2);
  float* proj          = (float*)alloc((size_t)T_ * H_ * 4);
  float* x2            = (float*)alloc((size_t)T_ * H_ * 4);
  unsigned short* F2   = (unsigned short*)alloc((size_t)T_ * KF_ * 2);
  float* h2            = (float*)alloc((size_t)T_ * H_ * 4);
  int*   sel           = (int*)alloc((size_t)T_ * 2 * 4);
  float* selw          = (float*)alloc((size_t)T_ * 2 * 4);
  float* T1 = qkvout;
  float* T2 = qkvout + (size_t)T_ * DFF_;
  float* Ye = Qb;
  unsigned short* F3 = F1;

  // ---- attention half ----
  k_conv_kan_w<<<(3 * H_ * H_ + 255) / 256, 256, 0, stream>>>(qkv_base, qkv_spl, qkv_scal, WB, 3 * H_, H_);
  k_rmsnorm_feat<<<T_, 256, 0, stream>>>(x, norm1_w, F1, nullptr);
  k_gemm_bf16<3 * H_, KF_><<<dim3((3 * H_) / 128, T_ / 128), 256, 0, stream>>>(F1, WB, qkvout);
  k_rope_split<<<(B_ * NH_ * S_ * (HD_ / 2) + 255) / 256, 256, 0, stream>>>(qkvout, rot_cos, rot_sin, Qb, Kb, Vb);
  k_attention<<<B_ * NH_ * S_, 64, 0, stream>>>(Qb, Kb, Vb, ctx);
  k_f32_to_bf16<<<(T_ * H_ + 255) / 256, 256, 0, stream>>>(ctx, ctxb, T_ * H_);
  k_conv_dense_T<<<(H_ * H_ + 255) / 256, 256, 0, stream>>>(out_w, WB, H_, H_);
  k_gemm_bf16<H_, H_><<<dim3(H_ / 128, T_ / 128), 256, 0, stream>>>(ctxb, WB, proj);
  k_add_bias_resid<<<(T_ * H_ + 255) / 256, 256, 0, stream>>>(x, proj, out_b, x2, T_ * H_);

  // ---- MoE KAN-FFN half ----
  k_rmsnorm_feat<<<T_, 256, 0, stream>>>(x2, norm2_w, F2, h2);
  k_gate<<<T_, 128, 0, stream>>>(h2, gate_w, sel, selw);
  k_copy<<<(T_ * H_ + 255) / 256, 256, 0, stream>>>(x2, out, T_ * H_);
  for (int e = 0; e < E_; ++e){
    const size_t wo  = (size_t)e * DFF_ * H_;
    const size_t wo3 = (size_t)e * H_ * DFF_;
    k_conv_kan_w<<<(DFF_ * H_ + 255) / 256, 256, 0, stream>>>(w1_base + wo, w1_spl + wo * NB_, w1_scal + wo, WB, DFF_, H_);
    k_gemm_bf16<DFF_, KF_><<<dim3(DFF_ / 128, T_ / 128), 256, 0, stream>>>(F2, WB, T1);
    k_conv_kan_w<<<(DFF_ * H_ + 255) / 256, 256, 0, stream>>>(w2_base + wo, w2_spl + wo * NB_, w2_scal + wo, WB, DFF_, H_);
    k_gemm_bf16<DFF_, KF_><<<dim3(DFF_ / 128, T_ / 128), 256, 0, stream>>>(F2, WB, T2);
    k_mul_feat<<<T_, 256, 0, stream>>>(T1, T2, F3);
    k_conv_kan_w<<<(H_ * DFF_ + 255) / 256, 256, 0, stream>>>(w3_base + wo3, w3_spl + wo3 * NB_, w3_scal + wo3, WB, H_, DFF_);
    k_gemm_bf16<H_, KF_><<<dim3(H_ / 128, T_ / 128), 256, 0, stream>>>(F3, WB, Ye);
    k_moe_accum<<<(T_ * H_ + 255) / 256, 256, 0, stream>>>(Ye, sel, selw, e, out, T_ * H_);
  }
}
